// Attention_54090818126522
// MI455X (gfx1250) — matrix operand order repaired and checked by reading, not compiled, not run on hardware
//
#include <hip/hip_runtime.h>
#include <hip/hip_bf16.h>

// Shapes fixed by the reference
#define BATCH   32
#define S_LEN   2048
#define ENC_D   1024
#define DEC_D   512
#define W_COLS  (DEC_D + ENC_D)   // 1536

#define TILE_S  256               // s-rows per workgroup (8 waves x 32)
#define NCHUNK  128               // N columns per accumulation pass (4 passes)
#define KSTEP   32                // K per bf16 WMMA
#define LDS_B_STRIDE 18           // dwords per staged B row (16 used + 2 pad, kills bank conflicts)

typedef __attribute__((ext_vector_type(16))) __bf16 v16bf;
typedef __attribute__((ext_vector_type(8)))  float  v8f;

// Pack two f32 into one dword of bf16 (truncating) with a single v_perm_b32.
__device__ __forceinline__ unsigned int pack_bf16(float lo, float hi) {
  return __builtin_amdgcn_perm(__builtin_bit_cast(unsigned int, hi),
                               __builtin_bit_cast(unsigned int, lo),
                               0x07060302u);
}

__device__ __forceinline__ float fast_tanh(float x) {
  // tanh(x) = (e^{2x}-1)/(e^{2x}+1) via v_exp_f32 / v_rcp_f32
  float xc = fminf(fmaxf(x, -9.0f), 9.0f);
  float e  = __builtin_amdgcn_exp2f(xc * 2.8853900817779268f); // 2*log2(e)
  return (e - 1.0f) * __builtin_amdgcn_rcpf(e + 1.0f);
}

// ---------------------------------------------------------------------------
// Kernel 1: hb[b,d] = bias[d] + sum_k hidden[b,k] * W[d, k]   (Wh = W[:, :512])
// ---------------------------------------------------------------------------
__global__ __launch_bounds__(512)
void attn_hproj_kernel(const float* __restrict__ hidden,
                       const float* __restrict__ W,
                       const float* __restrict__ bias,
                       float* __restrict__ hb) {
  __shared__ float lh[DEC_D];
  const int b = blockIdx.x;
  const int d = threadIdx.x;
  lh[d] = hidden[b * DEC_D + d];
  __syncthreads();

  const float4* w4 = (const float4*)(W + (size_t)d * W_COLS);
  const float4* h4 = (const float4*)lh;
  float acc = bias[d];
#pragma unroll 8
  for (int k = 0; k < DEC_D / 4; ++k) {
    float4 w = w4[k];
    float4 h = h4[k];
    acc = fmaf(w.x, h.x, acc);
    acc = fmaf(w.y, h.y, acc);
    acc = fmaf(w.z, h.z, acc);
    acc = fmaf(w.w, h.w, acc);
  }
  hb[b * DEC_D + d] = acc;
}

// ---------------------------------------------------------------------------
// Kernel 2: scores[b,s] = sum_d v[d] * tanh( (enc[b,s,:] @ We[d,:]) + hb[b,d] )
// WMMA bf16 GEMM. Each wave owns 2 M-tiles (32 s-rows) so every B fragment
// fetched from LDS feeds two back-to-back WMMAs (halves LDS bytes per WMMA,
// which is the binding per-wave resource). N done in 4 passes of 128 columns
// to keep accumulators at 128 VGPRs. A (encoder) and B (We) global loads for
// step k+1 are issued during step k's WMMA phase; B staged as bf16 in
// double-buffered LDS with one barrier per K-step.
// ---------------------------------------------------------------------------
__global__ __launch_bounds__(256)
void attn_energy_kernel(const float* __restrict__ enc,
                        const float* __restrict__ W,
                        const float* __restrict__ hb,
                        const float* __restrict__ v,
                        float* __restrict__ scores) {
  __shared__ unsigned int ldsB[2][NCHUNK * LDS_B_STRIDE]; // 2 x 9 KB bf16 B tiles
  __shared__ float ldsHB[DEC_D];                          // 2 KB
  __shared__ float ldsV[DEC_D];                           // 2 KB

  const int tid   = threadIdx.x;
  const int wave  = tid >> 5;
  const int lane  = tid & 31;
  const int b     = blockIdx.x >> 3;   // 8 s-blocks per batch
  const int sblk  = blockIdx.x & 7;
  const int sbase = sblk * TILE_S;

  // preload hb row (this batch) and v
  for (int i = tid; i < DEC_D; i += 256) {
    ldsHB[i] = hb[b * DEC_D + i];
    ldsV[i]  = v[i];
  }

  const int mrow  = lane & 15;   // M within tile (ISA A/B layout: lane%16)
  const int khalf = lane >> 4;   // lane-half selects K sub-range
  // two M-tiles per wave: rows [wave*32, wave*32+16) and [wave*32+16, wave*32+32)
  const float* __restrict__ arow0 =
      enc + (size_t)(b * S_LEN + sbase + wave * 32 + mrow) * ENC_D;
  const float* __restrict__ arow1 = arow0 + 16 * ENC_D;

  // B staging: thread t stages half a We row: row = t>>1, half = t&1 (16 floats)
  const int brow  = tid >> 1;
  const int bhalf = tid & 1;

  float partial0[8], partial1[8];
#pragma unroll
  for (int j = 0; j < 8; ++j) { partial0[j] = 0.0f; partial1[j] = 0.0f; }

  for (int npass = 0; npass < 4; ++npass) {
    const int nbase = npass * NCHUNK;
    const float* __restrict__ wsrc =
        W + (size_t)(nbase + brow) * W_COLS + DEC_D + bhalf * 16;

    v8f acc0[8], acc1[8];
#pragma unroll
    for (int t = 0; t < 8; ++t)
#pragma unroll
      for (int j = 0; j < 8; ++j) { acc0[t][j] = 0.0f; acc1[t][j] = 0.0f; }

    // ---- prologue: raw A for k0=0, stage B buf0 for k0=0 ----
    float4 a0r0, a0r1, a0r2, a0r3, a1r0, a1r1, a1r2, a1r3;
    a0r0 = *(const float4*)(arow0 + khalf * 8);
    a0r1 = *(const float4*)(arow0 + khalf * 8 + 4);
    a0r2 = *(const float4*)(arow0 + 16 + khalf * 8);
    a0r3 = *(const float4*)(arow0 + 16 + khalf * 8 + 4);
    a1r0 = *(const float4*)(arow1 + khalf * 8);
    a1r1 = *(const float4*)(arow1 + khalf * 8 + 4);
    a1r2 = *(const float4*)(arow1 + 16 + khalf * 8);
    a1r3 = *(const float4*)(arow1 + 16 + khalf * 8 + 4);
    {
      const float4* bsrc = (const float4*)wsrc;
      unsigned int* dst = &ldsB[0][brow * LDS_B_STRIDE + bhalf * 8];
#pragma unroll
      for (int i = 0; i < 4; ++i) {
        float4 f = bsrc[i];
        dst[2 * i]     = pack_bf16(f.x, f.y);
        dst[2 * i + 1] = pack_bf16(f.z, f.w);
      }
    }
    __syncthreads();

    int buf = 0;
    for (int k0 = 0; k0 < ENC_D; k0 += KSTEP) {
      const int  kn        = k0 + KSTEP;
      const bool have_next = (kn < ENC_D);

      // ---- issue next-step global loads up front (latency hidden by WMMAs) ----
      float4 an00, an01, an02, an03, an10, an11, an12, an13;
      float4 bnext[4];
      if (have_next) {
        an00 = *(const float4*)(arow0 + kn + khalf * 8);
        an01 = *(const float4*)(arow0 + kn + khalf * 8 + 4);
        an02 = *(const float4*)(arow0 + kn + 16 + khalf * 8);
        an03 = *(const float4*)(arow0 + kn + 16 + khalf * 8 + 4);
        an10 = *(const float4*)(arow1 + kn + khalf * 8);
        an11 = *(const float4*)(arow1 + kn + khalf * 8 + 4);
        an12 = *(const float4*)(arow1 + kn + 16 + khalf * 8);
        an13 = *(const float4*)(arow1 + kn + 16 + khalf * 8 + 4);
        const float4* bsrc = (const float4*)(wsrc + kn);
#pragma unroll
        for (int i = 0; i < 4; ++i) bnext[i] = bsrc[i];
        if (kn + KSTEP < ENC_D) {
          __builtin_prefetch(arow0 + kn + KSTEP + khalf * 8, 0, 1);
          __builtin_prefetch(arow1 + kn + KSTEP + khalf * 8, 0, 1);
        }
      }

      // ---- convert current A raw -> two WMMA A fragments (v_perm packing) ----
      union { v16bf v; unsigned int u[8]; } afrag0, afrag1;
      afrag0.u[0] = pack_bf16(a0r0.x, a0r0.y);
      afrag0.u[1] = pack_bf16(a0r0.z, a0r0.w);
      afrag0.u[2] = pack_bf16(a0r1.x, a0r1.y);
      afrag0.u[3] = pack_bf16(a0r1.z, a0r1.w);
      afrag0.u[4] = pack_bf16(a0r2.x, a0r2.y);
      afrag0.u[5] = pack_bf16(a0r2.z, a0r2.w);
      afrag0.u[6] = pack_bf16(a0r3.x, a0r3.y);
      afrag0.u[7] = pack_bf16(a0r3.z, a0r3.w);
      afrag1.u[0] = pack_bf16(a1r0.x, a1r0.y);
      afrag1.u[1] = pack_bf16(a1r0.z, a1r0.w);
      afrag1.u[2] = pack_bf16(a1r1.x, a1r1.y);
      afrag1.u[3] = pack_bf16(a1r1.z, a1r1.w);
      afrag1.u[4] = pack_bf16(a1r2.x, a1r2.y);
      afrag1.u[5] = pack_bf16(a1r2.z, a1r2.w);
      afrag1.u[6] = pack_bf16(a1r3.x, a1r3.y);
      afrag1.u[7] = pack_bf16(a1r3.z, a1r3.w);

      // ---- 8 B fragments x 2 WMMAs each: one LDS fetch feeds two tiles ----
      {
        const unsigned int* fbase =
            &ldsB[buf][mrow * LDS_B_STRIDE + khalf * 4];
#pragma unroll
        for (int nt = 0; nt < 8; ++nt) {
          union { v16bf v; unsigned int u[8]; } bfrag;
          const unsigned int* src = fbase + nt * 16 * LDS_B_STRIDE;
#pragma unroll
          for (int i = 0; i < 8; ++i) bfrag.u[i] = src[(i & 3) + ((i >> 2) << 3)];
          acc0[nt] = __builtin_amdgcn_wmma_f32_16x16x32_bf16(
              false, afrag0.v, false, bfrag.v, (short)0, acc0[nt], false, false);
          acc1[nt] = __builtin_amdgcn_wmma_f32_16x16x32_bf16(
              false, afrag1.v, false, bfrag.v, (short)0, acc1[nt], false, false);
        }
      }

      // ---- stage next B into the other buffer (loads have had time to land) ----
      if (have_next) {
        unsigned int* dst = &ldsB[buf ^ 1][brow * LDS_B_STRIDE + bhalf * 8];
#pragma unroll
        for (int i = 0; i < 4; ++i) {
          dst[2 * i]     = pack_bf16(bnext[i].x, bnext[i].y);
          dst[2 * i + 1] = pack_bf16(bnext[i].z, bnext[i].w);
        }
        a0r0 = an00; a0r1 = an01; a0r2 = an02; a0r3 = an03;
        a1r0 = an10; a1r1 = an11; a1r2 = an12; a1r3 = an13;
      }
      __syncthreads(); // staging complete + current buffer fully consumed
      buf ^= 1;
    } // k loop

    // ---- epilogue for this N pass: tanh + weighted partial sums ----
#pragma unroll
    for (int nt = 0; nt < 8; ++nt) {
      const int d  = nbase + nt * 16 + mrow; // C/D layout: lane%16 = N
      const float hv = ldsHB[d];
      const float vv = ldsV[d];
#pragma unroll
      for (int j = 0; j < 8; ++j) {
        partial0[j] = fmaf(vv, fast_tanh(acc0[nt][j] + hv), partial0[j]);
        partial1[j] = fmaf(vv, fast_tanh(acc1[nt][j] + hv), partial1[j]);
      }
    }
  } // npass

  // reduce over the 16-lane halves (lanes 0-15 -> rows j, lanes 16-31 -> rows j+8)
#pragma unroll
  for (int j = 0; j < 8; ++j) {
    float p0 = partial0[j];
    p0 += __shfl_xor(p0, 1, 32);
    p0 += __shfl_xor(p0, 2, 32);
    p0 += __shfl_xor(p0, 4, 32);
    p0 += __shfl_xor(p0, 8, 32);
    partial0[j] = p0;
    float p1 = partial1[j];
    p1 += __shfl_xor(p1, 1, 32);
    p1 += __shfl_xor(p1, 2, 32);
    p1 += __shfl_xor(p1, 4, 32);
    p1 += __shfl_xor(p1, 8, 32);
    partial1[j] = p1;
  }
  if ((lane & 15) == 0) {
    const int rbase = sbase + wave * 32 + khalf * 8;
#pragma unroll
    for (int j = 0; j < 8; ++j) {
      scores[b * S_LEN + rbase + j]      = partial0[j];
      scores[b * S_LEN + rbase + 16 + j] = partial1[j];
    }
  }
}

// ---------------------------------------------------------------------------
// Kernel 3: row softmax over S, writes final attn (B,1,S) to d_out
// ---------------------------------------------------------------------------
__global__ __launch_bounds__(256)
void attn_softmax_kernel(const float* __restrict__ scores,
                         float* __restrict__ out) {
  __shared__ float red[256];
  const int b   = blockIdx.x;
  const int tid = threadIdx.x;
  const float* row = scores + b * S_LEN;

  float vals[8];
  float mx = -3.4e38f;
#pragma unroll
  for (int i = 0; i < 8; ++i) {
    vals[i] = row[tid + i * 256];
    mx = fmaxf(mx, vals[i]);
  }
  red[tid] = mx;
  __syncthreads();
  for (int s = 128; s > 0; s >>= 1) {
    if (tid < s) red[tid] = fmaxf(red[tid], red[tid + s]);
    __syncthreads();
  }
  const float m = red[0];
  __syncthreads();

  float sum = 0.0f;
#pragma unroll
  for (int i = 0; i < 8; ++i) {
    vals[i] = __builtin_amdgcn_exp2f((vals[i] - m) * 1.4426950408889634f);
    sum += vals[i];
  }
  red[tid] = sum;
  __syncthreads();
  for (int s = 128; s > 0; s >>= 1) {
    if (tid < s) red[tid] += red[tid + s];
    __syncthreads();
  }
  const float inv = __builtin_amdgcn_rcpf(red[0]);
#pragma unroll
  for (int i = 0; i < 8; ++i)
    out[b * S_LEN + tid + i * 256] = vals[i] * inv;
}

// ---------------------------------------------------------------------------
extern "C" void kernel_launch(void* const* d_in, const int* in_sizes, int n_in,
                              void* d_out, int out_size, void* d_ws, size_t ws_size,
                              hipStream_t stream) {
  const float* hidden = (const float*)d_in[0]; // (32, 512)
  const float* enc    = (const float*)d_in[1]; // (32, 2048, 1024)
  const float* W      = (const float*)d_in[2]; // (512, 1536)
  const float* bias   = (const float*)d_in[3]; // (512,)
  const float* v      = (const float*)d_in[4]; // (512,)
  float*       out    = (float*)d_out;         // (32, 1, 2048)

  float* hbuf   = (float*)d_ws;                // 32*512 floats
  float* scores = hbuf + BATCH * DEC_D;        // 32*2048 floats

  attn_hproj_kernel<<<BATCH, 512, 0, stream>>>(hidden, W, bias, hbuf);
  attn_energy_kernel<<<BATCH * (S_LEN / TILE_S), 256, 0, stream>>>(enc, W, hbuf, v, scores);
  attn_softmax_kernel<<<BATCH, 256, 0, stream>>>(scores, out);
}
